// TimeLSTM_2963527434541
// MI455X (gfx1250) — matrix operand order repaired and checked by reading, not compiled, not run on hardware
//
#include <hip/hip_runtime.h>
#include <hip/hip_bf16.h>

typedef __attribute__((ext_vector_type(16))) _Float16 v16h;
typedef __attribute__((ext_vector_type(8)))  float    v8f;
typedef __attribute__((ext_vector_type(4)))  int      v4i;

#define B_DIM 2048
#define T_DIM 256
#define IN_DIM 65
#define H_DIM 256
#define G_DIM 1024   // 4*H
#define K_DIM 320    // 64 input features + 256 hidden
#define NKC   (K_DIM / 32)

// LDS row strides (padded)
#define LDA 40       // 32 + 8 halfs
#define LDB 264      // 256 + 8 halfs  (528 B, multiple of 16)
#define LDG 260      // 256 + 4 floats

// shared-memory layout: [As x2 | Bs x2], Gs aliases the whole region after k-loop
#define SM_A_BYTES (2 * 32 * LDA * 2)              // 5120
#define SM_B_BYTES (2 * 32 * LDB * 2)              // 33792
#define SM_BYTES   (SM_A_BYTES + SM_B_BYTES)       // 38912 (> 32*LDG*4 = 33280)

#if __has_builtin(__builtin_amdgcn_global_load_async_to_lds_b128) && \
    __has_builtin(__builtin_amdgcn_s_wait_asynccnt)
#define USE_ASYNC 1
#else
#define USE_ASYNC 0
#endif

// addrspace-qualified v4i pointers for the async builtin
#define ASYNC_G(p) ((__attribute__((address_space(1))) v4i*) \
                    (__attribute__((address_space(1))) void*)(p))
#define ASYNC_L(p) ((__attribute__((address_space(3))) v4i*) \
                    (__attribute__((address_space(3))) void*)(p))

__device__ __forceinline__ float sigmoidf_(float x) {
    return 1.0f / (1.0f + __expf(-x));
}

// ---------------------------------------------------------------------------
// Prep: interleaved f16 weights BigB[k][4*j+g] (k<64 -> W_ih, k>=64 -> W_hh),
// combined bias biasI[4*j+g] = b_ih+b_hh, zero h0 and c.
// ---------------------------------------------------------------------------
__global__ void tlstm_prep(const float* __restrict__ W_ih,
                           const float* __restrict__ W_hh,
                           const float* __restrict__ b_ih,
                           const float* __restrict__ b_hh,
                           _Float16* __restrict__ BigB,
                           float* __restrict__ biasI,
                           float* __restrict__ h0,
                           float* __restrict__ cbuf)
{
    long e = (long)blockIdx.x * blockDim.x + threadIdx.x;
    const long NB = (long)K_DIM * G_DIM;          // 327680
    if (e < NB) {
        int k   = (int)(e / G_DIM);
        int col = (int)(e % G_DIM);
        int j = col >> 2, g = col & 3;
        int row = g * H_DIM + j;                  // row in original (4H, K) weights
        float v = (k < 64) ? W_ih[row * 64 + k]
                           : W_hh[row * H_DIM + (k - 64)];
        BigB[e] = (_Float16)v;
        return;
    }
    e -= NB;
    if (e < G_DIM) {
        int col = (int)e;
        int j = col >> 2, g = col & 3;
        biasI[col] = b_ih[g * H_DIM + j] + b_hh[g * H_DIM + j];
        return;
    }
    e -= G_DIM;
    const long NH = (long)B_DIM * H_DIM;
    if (e < NH) { h0[e] = 0.0f; return; }
    e -= NH;
    if (e < NH) { cbuf[e] = 0.0f; return; }
}

// ---------------------------------------------------------------------------
// One timestep: gates = [x_t | h] (2048x320) @ BigB (320x1024) via WMMA with
// a double-buffered async-to-LDS pipeline, then fused LSTM cell update.
// Grid: (64, 4) blocks of 256 threads. Block tile: M=32, N'=256 (64 hidden x
// 4 interleaved gates).
// ---------------------------------------------------------------------------
__global__ __launch_bounds__(256)
void tlstm_step(const float* __restrict__ x,
                const float* __restrict__ h_in,
                float* __restrict__ h_out,
                float* __restrict__ cbuf,
                const _Float16* __restrict__ BigB,
                const float* __restrict__ biasI,
                const float* __restrict__ Wt,
                const float* __restrict__ bt,
                int t)
{
    __shared__ __align__(16) char smem[SM_BYTES];
    _Float16* As = (_Float16*)(smem);              // 2 x [32][LDA]
    _Float16* Bs = (_Float16*)(smem + SM_A_BYTES); // 2 x [32][LDB]
    float*    Gs = (float*)smem;                   // [32][LDG], aliases As/Bs

    const int tid  = threadIdx.x;
    const int lane = tid & 31;
    const int w    = tid >> 5;          // wave 0..7
    const int wm   = w & 1;             // m-tile of wave (0..1)
    const int wn   = w >> 1;            // n-group of wave (0..3)
    const int bm0  = blockIdx.x * 32;   // batch-row base
    const int bn0  = blockIdx.y * 256;  // n' base (4*j+g space)

    v8f acc[4];
    #pragma unroll
    for (int i = 0; i < 4; ++i)
        #pragma unroll
        for (int q = 0; q < 8; ++q) acc[i][q] = 0.0f;

    float a_reg[4];

    // --- A chunk: global loads into registers (issue early, use late) ---
    auto loadA = [&](int kc) {
        const int k0 = kc * 32;
        #pragma unroll
        for (int i = 0; i < 4; ++i) {
            int e   = tid + i * 256;
            int row = e >> 5, col = e & 31;
            int bg  = bm0 + row;
            int k   = k0 + col;
            a_reg[i] = (k < 64) ? x[((size_t)bg * T_DIM + t) * IN_DIM + 1 + k]
                                : h_in[bg * H_DIM + (k - 64)];
        }
    };
    // --- A chunk: f32->f16 convert + LDS store ---
    auto storeA = [&](int buf) {
        _Float16* dst = As + buf * 32 * LDA;
        #pragma unroll
        for (int i = 0; i < 4; ++i) {
            int e   = tid + i * 256;
            int row = e >> 5, col = e & 31;
            dst[row * LDA + col] = (_Float16)a_reg[i];
        }
    };
    // --- B chunk 32x256 f16: async global->LDS (CDNA5) or sync fallback ---
    auto issueB = [&](int kc, int buf) {
        const _Float16* gsrc = BigB + (size_t)kc * 32 * G_DIM + bn0;
        _Float16*       ldst = Bs + buf * 32 * LDB;
#if USE_ASYNC
        #pragma unroll
        for (int i = 0; i < 4; ++i) {
            int e   = tid + i * 256;       // 1024 x b128 total
            int row = e >> 5, q = e & 31;  // 32 x 16B per row
            __builtin_amdgcn_global_load_async_to_lds_b128(
                ASYNC_G(gsrc + row * G_DIM + q * 8),
                ASYNC_L(ldst + row * LDB + q * 8),
                0, 0);
        }
#else
        const uint4* src = (const uint4*)gsrc;
        #pragma unroll
        for (int i = 0; i < 4; ++i) {
            int e   = tid + i * 256;
            int row = e >> 5, q = e & 31;
            uint4 d = src[row * (G_DIM / 8) + q];
            *(uint4*)&ldst[row * LDB + q * 8] = d;
        }
        if ((kc + 1) * 32 < K_DIM)
            __builtin_prefetch(BigB + (size_t)(kc + 1) * 32 * G_DIM + bn0, 0, 3);
#endif
    };
    // --- one K-chunk of WMMAs ---
    auto computeChunk = [&](int buf) {
        const _Float16* Ab = As + buf * 32 * LDA;
        const _Float16* Bb = Bs + buf * 32 * LDB;
        v16h afrag;   // lane = M (mod 16), VGPR/lane-half = K
        {
            int m    = wm * 16 + (lane & 15);
            int kadd = (lane >> 4) * 8;
            #pragma unroll
            for (int v = 0; v < 8; ++v) {
                int k = ((v & 4) ? 16 : 0) + kadd + (v & 3) * 2;
                afrag[2 * v]     = Ab[m * LDA + k];
                afrag[2 * v + 1] = Ab[m * LDA + k + 1];
            }
        }
        #pragma unroll
        for (int tt = 0; tt < 4; ++tt) {
            int n0 = wn * 64 + tt * 16;
            v16h bfrag;   // lane = N (mod 16), VGPR/lane-half = K
            #pragma unroll
            for (int v = 0; v < 8; ++v) {
                int k = ((v & 4) ? 16 : 0) + (lane >> 4) * 8 + (v & 3) * 2;
                bfrag[2 * v]     = Bb[k * LDB + n0 + (lane & 15)];
                bfrag[2 * v + 1] = Bb[(k + 1) * LDB + n0 + (lane & 15)];
            }
            acc[tt] = __builtin_amdgcn_wmma_f32_16x16x32_f16(
                false, afrag, false, bfrag, (short)0, acc[tt], false, false);
        }
    };

    // ---- pipelined K loop: copy(kc+1) overlaps compute(kc), 1 barrier/iter --
    loadA(0); storeA(0);
    issueB(0, 0);
#if USE_ASYNC
    __builtin_amdgcn_s_wait_asynccnt(0);
#endif
    __syncthreads();

    for (int kc = 0; kc < NKC; ++kc) {
        const int cur = kc & 1, nxt = cur ^ 1;
        const bool have_next = (kc + 1 < NKC);
        if (have_next) { issueB(kc + 1, nxt); loadA(kc + 1); }
        computeChunk(cur);                 // async B(kc+1) in flight meanwhile
        if (have_next) storeA(nxt);
#if USE_ASYNC
        __builtin_amdgcn_s_wait_asynccnt(0);
#endif
        __syncthreads();
    }

    // ---- spill accumulators to LDS (Gs aliases A/B buffers, now dead) ----
    #pragma unroll
    for (int tt = 0; tt < 4; ++tt) {
        int n  = wn * 64 + tt * 16 + (lane & 15);
        int mb = wm * 16 + (lane >> 4) * 8;
        #pragma unroll
        for (int r = 0; r < 8; ++r)
            Gs[(mb + r) * LDG + n] = acc[tt][r];
    }
    __syncthreads();

    // ---- fused LSTM cell update: 32 rows x 64 hidden, 8 items/thread ----
    #pragma unroll
    for (int i = 0; i < 8; ++i) {
        int e  = tid + i * 256;      // 0..2047
        int bl = e >> 6;             // batch row 0..31
        int jl = e & 63;             // hidden unit 0..63
        int bg = bm0 + bl;
        int jg = blockIdx.y * 64 + jl;
        float gi = Gs[bl * LDG + 4 * jl + 0] + biasI[4 * jg + 0];
        float gf = Gs[bl * LDG + 4 * jl + 1] + biasI[4 * jg + 1];
        float gg = Gs[bl * LDG + 4 * jl + 2] + biasI[4 * jg + 2];
        float go = Gs[bl * LDG + 4 * jl + 3] + biasI[4 * jg + 3];
        float td = x[((size_t)bg * T_DIM + t) * IN_DIM];     // time diff
        float d  = sigmoidf_(td * Wt[jg] + bt[jg]);
        float c  = cbuf[bg * H_DIM + jg] * d;
        c = sigmoidf_(gf) * c + sigmoidf_(gi) * tanhf(gg);
        float h = sigmoidf_(go) * tanhf(c);
        cbuf[bg * H_DIM + jg]  = c;
        h_out[bg * H_DIM + jg] = h;
    }
}

// ---------------------------------------------------------------------------
// Final projection: out[b] = h_T[b,:] . Wf + bf   (one wave per batch row)
// ---------------------------------------------------------------------------
__global__ __launch_bounds__(256)
void tlstm_final(const float* __restrict__ h, const float* __restrict__ Wf,
                 const float* __restrict__ bf, float* __restrict__ out)
{
    __shared__ float red[256];
    int tid = threadIdx.x, lane = tid & 31, w = tid >> 5;
    int b = blockIdx.x * 8 + w;
    float p = 0.0f;
    #pragma unroll
    for (int s = 0; s < 8; ++s) {
        int j = lane + s * 32;
        p += h[b * H_DIM + j] * Wf[j];
    }
    red[tid] = p;
    __syncthreads();
    if (lane == 0) {
        float s = 0.0f;
        #pragma unroll
        for (int i = 0; i < 32; ++i) s += red[w * 32 + i];
        out[b] = s + bf[0];
    }
}

// ---------------------------------------------------------------------------
extern "C" void kernel_launch(void* const* d_in, const int* in_sizes, int n_in,
                              void* d_out, int out_size, void* d_ws, size_t ws_size,
                              hipStream_t stream)
{
    const float* x    = (const float*)d_in[0];
    const float* W_ih = (const float*)d_in[1];
    const float* W_hh = (const float*)d_in[2];
    const float* b_ih = (const float*)d_in[3];
    const float* b_hh = (const float*)d_in[4];
    const float* Wt   = (const float*)d_in[5];
    const float* bt   = (const float*)d_in[6];
    const float* Wf   = (const float*)d_in[7];
    const float* bf   = (const float*)d_in[8];
    float* out = (float*)d_out;

    const size_t NH = (size_t)B_DIM * H_DIM;        // 524288
    float*    h0    = (float*)d_ws;                 // 2 MB
    float*    h1    = h0 + NH;                      // 2 MB
    float*    cbuf  = h1 + NH;                      // 2 MB
    _Float16* BigB  = (_Float16*)(cbuf + NH);       // 640 KB f16
    float*    biasI = (float*)(BigB + (size_t)K_DIM * G_DIM);  // 4 KB

    {
        long total = (long)K_DIM * G_DIM + G_DIM + 2 * (long)NH;
        int blocks = (int)((total + 255) / 256);
        tlstm_prep<<<blocks, 256, 0, stream>>>(W_ih, W_hh, b_ih, b_hh,
                                               BigB, biasI, h0, cbuf);
    }
    float* hb[2] = { h0, h1 };
    for (int t = 0; t < T_DIM; ++t) {
        tlstm_step<<<dim3(64, 4), 256, 0, stream>>>(
            x, hb[t & 1], hb[(t + 1) & 1], cbuf, BigB, biasI, Wt, bt, t);
    }
    // T even -> final h lives in hb[0]
    tlstm_final<<<B_DIM / 8, 256, 0, stream>>>(h0, Wf, bf, out);
}
